// FusionMamba_15977278341332
// MI455X (gfx1250) — matrix operand order repaired and checked by reading, not compiled, not run on hardware
//
#include <hip/hip_runtime.h>

// ---------------------------------------------------------------------------
// FusionMamba forward for MI455X (gfx1250, wave32, WMMA + TDM).
// GEMMs: operands pre-staged as zero-padded bf16; guard-free WMMA kernel
// templated on <K, ACT>: 4x global_load_b128 + v_wmma_f32_16x16x32_bf16 per
// K-step, fully unrolled, activation inlined per instantiation.
// Selective scan: both directions in one launch (gridDim.y=2), h[16] in VGPRs;
// B_t/C_t chunks (16x64 f32 tiles, row stride nx) are DMA'd into LDS by the
// Tensor Data Mover (tensor_load_to_lds) with double buffering so the next
// chunk's DMA overlaps the current chunk's recurrence; completion ordered with
// s_wait_tensorcnt. silu(z) gating fused into the scan epilogue.
// ---------------------------------------------------------------------------

typedef __attribute__((ext_vector_type(16))) __bf16       v16bf;
typedef __attribute__((ext_vector_type(8)))  __bf16       bf16x8;
typedef __attribute__((ext_vector_type(8)))  float        v8f;
typedef __attribute__((ext_vector_type(4)))  unsigned int u32x4;
typedef __attribute__((ext_vector_type(8)))  int          i32x8;
typedef __attribute__((ext_vector_type(4)))  int          i32x4;

#if __has_builtin(__builtin_amdgcn_tensor_load_to_lds) && \
    __has_builtin(__builtin_amdgcn_s_wait_tensorcnt)
#define HAVE_TDM 1
#else
#define HAVE_TDM 0
#endif

#define DEVINL __device__ __forceinline__

constexpr int BN = 2;            // batch
constexpr int CD = 64;           // channels (DIM)
constexpr int HH = 64, WW = 64;
constexpr int LL = HH * WW;      // 4096 sequence length
constexpr int SS = BN * LL;      // 8192 rows
constexpr int NST = 16;          // N_STATE

// ---------------------------- device helpers -------------------------------

DEVINL float nan2num(float x) {
  if (__builtin_isnan(x)) return 0.f;
  if (__builtin_isinf(x)) return x > 0.f ? 1.f : -1.f;
  return x;
}
DEVINL float siluf(float x) { return x / (1.f + __expf(-x)); }

DEVINL __bf16 f2bf(float x) {
  unsigned u = __builtin_bit_cast(unsigned, x);
  u += 0x7fffu + ((u >> 16) & 1u);           // round-to-nearest-even
  unsigned short h = (unsigned short)(u >> 16);
  return __builtin_bit_cast(__bf16, h);
}

template <int ACT>
DEVINL float act_f(float v) {
  if constexpr (ACT == 1) return siluf(v);
  else if constexpr (ACT == 2) return 1.f / (1.f + __expf(-v));
  else if constexpr (ACT == 3) return 0.5f * v * (1.f + erff(v * 0.70710678f));
  else if constexpr (ACT == 4)   // softplus via hw exp/log, numerically stable
    return fmaxf(v, 0.f) + __logf(1.f + __expf(-fabsf(v)));
  else return v;
}

// ------------------------------- kernels -----------------------------------

// NCHW -> (B, L, C) with nan_to_num
__global__ void k_prep(const float* __restrict__ img, float* __restrict__ seq) {
  int idx = blockIdx.x * blockDim.x + threadIdx.x;
  if (idx >= BN * CD * LL) return;
  int b = idx / (CD * LL);
  int c = (idx / LL) % CD;
  int p = idx % LL;
  seq[(b * LL + p) * CD + c] = nan2num(img[idx]);
}

// Strided fp32 -> dense zero-padded bf16 (rowsP x colsP), for GEMM staging.
__global__ void k_cvt(const float* __restrict__ x, int ldx, int offx,
                      int rows, int cols, __bf16* __restrict__ y,
                      int colsP, int rowsP) {
  int idx = blockIdx.x * blockDim.x + threadIdx.x;
  if (idx >= rowsP * colsP) return;
  int r = idx / colsP, c = idx % colsP;
  float v = (r < rows && c < cols) ? x[r * ldx + offx + c] : 0.f;
  y[idx] = f2bf(v);
}

// LayerNorm over last dim C (C multiple of 32, <=128). One wave32 per row.
__global__ void k_layernorm(const float* __restrict__ x, const float* __restrict__ g,
                            const float* __restrict__ bta, float* __restrict__ y,
                            int rows, int C) {
  int row  = (blockIdx.x * blockDim.x + threadIdx.x) >> 5;
  int lane = threadIdx.x & 31;
  if (row >= rows) return;
  float vv[4];
  int n = C >> 5;
  float s = 0.f, s2 = 0.f;
  for (int i = 0; i < n; ++i) {
    float v = x[row * C + lane + (i << 5)];
    vv[i] = v; s += v; s2 += v * v;
  }
  for (int off = 16; off > 0; off >>= 1) {
    s  += __shfl_xor(s,  off, 32);
    s2 += __shfl_xor(s2, off, 32);
  }
  float mean = s / (float)C;
  float var  = s2 / (float)C - mean * mean;
  float rs   = rsqrtf(var + 1e-5f);
  for (int i = 0; i < n; ++i) {
    int c = lane + (i << 5);
    y[row * C + c] = (vv[i] - mean) * rs * g[c] + bta[c];
  }
}

// out[M,N] = act(A[M,KP] @ W[N,KP]^T + bias). A, W dense bf16, K/N padded:
// no guards on loads; per K-step 4x b128 loads + 1 WMMA. 16x16 tile per wave.
template <int KP, int ACT>
__global__ void k_gemm_fast(const __bf16* __restrict__ A, const __bf16* __restrict__ W,
                            const float* __restrict__ bias, float* __restrict__ out,
                            int ldo, int offo, int M, int N) {
  int wave = blockIdx.x * (blockDim.x >> 5) + (threadIdx.x >> 5);
  int lane = threadIdx.x & 31;
  int tilesN = (N + 15) >> 4;
  if (wave >= (M >> 4) * tilesN) return;       // wave-uniform
  int m0 = (wave / tilesN) << 4;
  int n0 = (wave % tilesN) << 4;
  int khi = (lane >> 4) & 1;

  const __bf16* Ap = A + (size_t)(m0 + (lane & 15)) * KP + 8 * khi;
  const __bf16* Wp = W + (size_t)(n0 + (lane & 15)) * KP + 8 * khi;

  v8f acc = {};
#pragma unroll
  for (int kk = 0; kk < KP; kk += 32) {
    bf16x8 a0 = *(const bf16x8*)(Ap + kk);
    bf16x8 a1 = *(const bf16x8*)(Ap + kk + 16);
    bf16x8 b0 = *(const bf16x8*)(Wp + kk);
    bf16x8 b1 = *(const bf16x8*)(Wp + kk + 16);
    v16bf a = __builtin_shufflevector(a0, a1, 0, 1, 2, 3, 4, 5, 6, 7,
                                      8, 9, 10, 11, 12, 13, 14, 15);
    v16bf b = __builtin_shufflevector(b0, b1, 0, 1, 2, 3, 4, 5, 6, 7,
                                      8, 9, 10, 11, 12, 13, 14, 15);
    acc = __builtin_amdgcn_wmma_f32_16x16x32_bf16(false, a, false, b, (short)0,
                                                  acc, false, false);
  }

  int col   = n0 + (lane & 15);
  int rbase = m0 + ((lane < 16) ? 0 : 8);
  if (col < N) {
    float bv = bias ? bias[col] : 0.f;
#pragma unroll
    for (int i = 0; i < 8; ++i)
      out[(rbase + i) * ldo + offo + col] = act_f<ACT>(acc[i] + bv);
  }
}

// y[b,t,c] = x[b, rev? L-1-t : t, c]  (slice / concat / flip / copy)
__global__ void k_slice(const float* __restrict__ x, int ldx, int offx,
                        float* __restrict__ y, int ldy, int offy,
                        int d, int rev) {
  int idx = blockIdx.x * blockDim.x + threadIdx.x;
  if (idx >= BN * LL * d) return;
  int b = idx / (LL * d);
  int t = (idx / d) % LL;
  int c = idx % d;
  int ts = rev ? (LL - 1 - t) : t;
  y[(b * LL + t) * ldy + offy + c] = x[(b * LL + ts) * ldx + offx + c];
}

// causal depthwise conv (K=4) + SiLU, dense (B,L,d)
__global__ void k_dwconv_silu(const float* __restrict__ x, const float* __restrict__ w,
                              const float* __restrict__ bias, float* __restrict__ y,
                              int d) {
  int idx = blockIdx.x * blockDim.x + threadIdx.x;
  if (idx >= BN * LL * d) return;
  int b = idx / (LL * d);
  int t = (idx / d) % LL;
  int c = idx % d;
  float s = bias[c];
#pragma unroll
  for (int k = 0; k < 4; ++k) {
    int tt = t - 3 + k;
    if (tt >= 0) s += x[(b * LL + tt) * d + c] * w[c * 4 + k];
  }
  y[idx] = siluf(s);
}

#if HAVE_TDM
// Issue one TDM 2D-tile load: 16 x 64 fp32 tile, row stride strideElems,
// global -> LDS. D# layout per CDNA5 ISA 8.3/8.4 (2D: groups 2/3 zero).
DEVINL void tdm_load_tile(unsigned ldsOff, const float* g, int strideElems) {
  unsigned long long ga = (unsigned long long)(uintptr_t)g;
  u32x4 g0;
  g0[0] = 1u;                                   // count=1, user descriptor
  g0[1] = ldsOff;                               // lds_addr (bytes)
  g0[2] = (unsigned)ga;                         // global_addr[31:0]
  g0[3] = (unsigned)(ga >> 32) | (2u << 30);    // global_addr[56:32] | type=2
  i32x8 g1;
  g1[0] = (int)(2u << 16);                      // workgroup_mask=0, data_size=4B
  g1[1] = (int)(16u << 16);                     // tensor_dim0 = 16
  g1[2] = (int)(64u << 16);                     // tensor_dim1 = 64
  g1[3] = (int)(16u << 16);                     // tile_dim0 = 16
  g1[4] = 64;                                   // tile_dim1 = 64, tile_dim2 = 0
  g1[5] = strideElems;                          // tensor_dim0_stride
  g1[6] = 0;
  g1[7] = 0;                                    // tensor_dim1_stride unused (2D)
  i32x4 z4 = {0, 0, 0, 0};
  i32x8 z8 = {0, 0, 0, 0, 0, 0, 0, 0};
  __builtin_amdgcn_tensor_load_to_lds(g0, g1, z4, z4, z8, 0);
}
#endif

// Selective scan, both directions in one launch. grid=(B,2); block=d threads.
struct ScanArgs {
  const float* u[2];
  const float* dt[2];
  const float* xdbl[2];
  const float* Alog[2];
  const float* Dp[2];
  float*       y[2];
};
__global__ void k_scan(ScanArgs sa, const float* __restrict__ z, int ldz, int offz,
                       int ldbc, int offB, int offC, int d) {
  __shared__ float Bs[2][64 * NST];
  __shared__ float Cs[2][64 * NST];
  const int b   = blockIdx.x;
  const int dir = blockIdx.y;
  const int c   = threadIdx.x;
  const float* __restrict__ u    = sa.u[dir];
  const float* __restrict__ dt   = sa.dt[dir];
  const float* __restrict__ xdbl = sa.xdbl[dir];
  float*       __restrict__ y    = sa.y[dir];

  float Arow[NST], h[NST];
#pragma unroll
  for (int n = 0; n < NST; ++n) {
    Arow[n] = -__expf(sa.Alog[dir][c * NST + n]);
    h[n] = 0.f;
  }
  float Dv = sa.Dp[dir][c];

#if HAVE_TDM
  const bool issuer = (threadIdx.x < 32);       // wave 0 (wave-uniform branch)
  if (issuer) {
    const float* gB = &xdbl[(size_t)(b * LL) * ldbc + offB];
    const float* gC = &xdbl[(size_t)(b * LL) * ldbc + offC];
    tdm_load_tile((unsigned)(uintptr_t)&Bs[0][0], gB, ldbc);
    tdm_load_tile((unsigned)(uintptr_t)&Cs[0][0], gC, ldbc);
  }
  for (int t0 = 0; t0 < LL; t0 += 64) {
    int cur = (t0 >> 6) & 1;
    __syncthreads();                            // everyone done with buf cur^1
    if (t0 + 64 < LL) {
      if (issuer) {
        const float* gB = &xdbl[(size_t)(b * LL + t0 + 64) * ldbc + offB];
        const float* gC = &xdbl[(size_t)(b * LL + t0 + 64) * ldbc + offC];
        tdm_load_tile((unsigned)(uintptr_t)&Bs[cur ^ 1][0], gB, ldbc);
        tdm_load_tile((unsigned)(uintptr_t)&Cs[cur ^ 1][0], gC, ldbc);
        __builtin_amdgcn_s_wait_tensorcnt(2);   // current chunk landed (in-order)
      }
    } else if (issuer) {
      __builtin_amdgcn_s_wait_tensorcnt(0);
    }
    __syncthreads();                            // chunk `cur` visible to all
#else
  for (int t0 = 0; t0 < LL; t0 += 64) {
    int cur = 0;
    __syncthreads();
    for (int i = c; i < 64 * NST; i += d) {
      int t = i >> 4, n = i & 15;
      int g = (b * LL + t0 + t) * ldbc;
      Bs[0][i] = xdbl[g + offB + n];
      Cs[0][i] = xdbl[g + offC + n];
    }
    __syncthreads();
#endif
    for (int t = 0; t < 64; ++t) {
      int gi = (b * LL + t0 + t) * d + c;
      float dtv = dt[gi];
      float uv  = u[gi];
      float acc = 0.f;
#pragma unroll
      for (int n = 0; n < NST; ++n) {
        float dA = __expf(dtv * Arow[n]);
        float hv = dA * h[n] + (dtv * Bs[cur][t * NST + n]) * uv;
        h[n] = hv;
        acc += hv * Cs[cur][t * NST + n];
      }
      int tt = t0 + t;
      int to = dir ? (LL - 1 - tt) : tt;            // original orientation
      float zv = z[(b * LL + to) * ldz + offz + c]; // fused silu(z) gate
      y[(b * LL + to) * d + c] = (acc + uv * Dv) * siluf(zv);
    }
  }
}

// out = (a + b) * s
__global__ void k_scale_add(float* __restrict__ out, const float* __restrict__ a,
                            const float* __restrict__ b, int n, float s) {
  int i = blockIdx.x * blockDim.x + threadIdx.x;
  if (i < n) out[i] = (a[i] + b[i]) * s;
}

__global__ void k_sigclip(const float* __restrict__ x, float* __restrict__ y, int n) {
  int i = blockIdx.x * blockDim.x + threadIdx.x;
  if (i < n) {
    float v = 1.f / (1.f + __expf(-x[i]));
    y[i] = fminf(fmaxf(v, 0.01f), 0.99f);
  }
}

__global__ void k_mix(float* __restrict__ out, const float* __restrict__ pb,
                      const float* __restrict__ wgt, const float* __restrict__ n0,
                      const float* __restrict__ skip, int n) {
  int i = blockIdx.x * blockDim.x + threadIdx.x;
  if (i < n) {
    float w = wgt[i];
    out[i] = nan2num(pb[i]) * w + n0[i] * (1.f - w) + skip[i];
  }
}

__global__ void k_fuse(const float* __restrict__ a, const float* __restrict__ b,
                       const float* __restrict__ alpha, const float* __restrict__ beta,
                       float* __restrict__ out, int n) {
  int i = blockIdx.x * blockDim.x + threadIdx.x;
  if (i < n) {
    float al = fminf(fmaxf(alpha[0], 0.1f), 2.f);
    float be = fminf(fmaxf(beta[0],  0.1f), 2.f);
    out[i] = (al * a[i] + be * b[i]) / (al + be + 1e-6f);
  }
}

__global__ void k_final(const float* __restrict__ fuse, const float* __restrict__ x1,
                        const float* __restrict__ x2, float* __restrict__ out) {
  int idx = blockIdx.x * blockDim.x + threadIdx.x;
  if (idx >= BN * CD * LL) return;
  int b = idx / (CD * LL);
  int c = (idx / LL) % CD;
  int l = idx % LL;
  float f = nan2num(fuse[(b * LL + l) * CD + c]);
  out[idx]                = (x1[(b * LL + l) * CD + c] + f) * 0.5f;
  out[BN * CD * LL + idx] = (x2[(b * LL + l) * CD + c] + f) * 0.5f;
}

// ------------------------------ host side ----------------------------------

namespace {

struct DirP   { const float *conv_w, *conv_b, *xp_w, *dt_w, *dt_b, *A_log, *D; };
struct MambaP { const float* in_w; DirP fwd, bwd; const float *ng, *nb, *out_w; };
struct CrossP { const float *n0g, *n0b, *n1g, *n1b; MambaP m;
                const float *png, *pnb, *pb_w, *pb_b, *cw_w, *cw_b, *cwg, *cwb; };
struct SingleP{ const float *ng0, *nb0; MambaP m; const float *png, *pnb; };

struct Cursor {
  void* const* in; int i;
  const float* next() { return reinterpret_cast<const float*>(in[i++]); }
};
DirP read_dir(Cursor& c) {
  DirP p; p.conv_w=c.next(); p.conv_b=c.next(); p.xp_w=c.next(); p.dt_w=c.next();
  p.dt_b=c.next(); p.A_log=c.next(); p.D=c.next(); return p;
}
MambaP read_mamba(Cursor& c) {
  MambaP m; m.in_w=c.next(); m.fwd=read_dir(c); m.bwd=read_dir(c);
  m.ng=c.next(); m.nb=c.next(); m.out_w=c.next(); return m;
}
CrossP read_cross(Cursor& c) {
  CrossP p; p.n0g=c.next(); p.n0b=c.next(); p.n1g=c.next(); p.n1b=c.next();
  p.m=read_mamba(c); p.png=c.next(); p.pnb=c.next(); p.pb_w=c.next(); p.pb_b=c.next();
  p.cw_w=c.next(); p.cw_b=c.next(); p.cwg=c.next(); p.cwb=c.next(); return p;
}
SingleP read_single(Cursor& c) {
  SingleP p; p.ng0=c.next(); p.nb0=c.next(); p.m=read_mamba(c);
  p.png=c.next(); p.pnb=c.next(); return p;
}

struct WS {
  float *n0, *n1, *comb, *t64a, *t64b, *wgt;
  float *xz, *xdir, *ysum, *ln, *mo, *fuse;
  float *xc[2], *xdbl[2], *dt[2], *ydir[2];
  __bf16 *Abf, *Wbf;
};

inline int eb(int n) { return (n + 255) / 256; }

template <int KP>
void launch_gemm_k(hipStream_t s, int blocks, const __bf16* A, const __bf16* Wt,
                   const float* bias, float* out, int ldo, int offo,
                   int M, int N, int act) {
  if (act == 3)
    k_gemm_fast<KP, 3><<<blocks, 256, 0, s>>>(A, Wt, bias, out, ldo, offo, M, N);
  else if (act == 4)
    k_gemm_fast<KP, 4><<<blocks, 256, 0, s>>>(A, Wt, bias, out, ldo, offo, M, N);
  else
    k_gemm_fast<KP, 0><<<blocks, 256, 0, s>>>(A, Wt, bias, out, ldo, offo, M, N);
}

void gemm(hipStream_t s, const float* A, int lda, int offa, const float* W,
          const float* bias, float* out, int ldo, int offo,
          int M, int N, int K, int act, WS& w) {
  int Kp = (K + 31) & ~31;
  int Np = (N + 15) & ~15;
  k_cvt<<<eb(M * Kp),  256, 0, s>>>(A, lda, offa, M, K, w.Abf, Kp, M);
  k_cvt<<<eb(Np * Kp), 256, 0, s>>>(W, K,   0,    N, K, w.Wbf, Kp, Np);
  int tiles  = (M >> 4) * ((N + 15) >> 4);
  int blocks = (tiles + 7) / 8;
  switch (Kp) {
    case 32:  launch_gemm_k<32> (s, blocks, w.Abf, w.Wbf, bias, out, ldo, offo, M, N, act); break;
    case 64:  launch_gemm_k<64> (s, blocks, w.Abf, w.Wbf, bias, out, ldo, offo, M, N, act); break;
    case 128: launch_gemm_k<128>(s, blocks, w.Abf, w.Wbf, bias, out, ldo, offo, M, N, act); break;
    default:  launch_gemm_k<256>(s, blocks, w.Abf, w.Wbf, bias, out, ldo, offo, M, N, act); break;
  }
}
void lnorm(hipStream_t s, const float* x, const float* g, const float* b,
           float* y, int C) {
  k_layernorm<<<eb(SS * 32), 256, 0, s>>>(x, g, b, y, SS, C);
}

void run_mamba(hipStream_t s, const MambaP& m, const float* x, int ldx, int offx,
               int d, float* out, WS& w) {
  const int d2 = 2 * d;
  const int r  = (d + 15) / 16;          // dt_rank
  const int nx = r + 2 * NST;

  gemm(s, x, ldx, offx, m.in_w, nullptr, w.xz, d2, 0, SS, d2, d, 0, w);

  for (int dir = 0; dir < 2; ++dir) {
    const DirP& dp = dir ? m.bwd : m.fwd;
    k_slice<<<eb(SS * d), 256, 0, s>>>(w.xz, d2, 0, w.xdir, d, 0, d, dir);
    k_dwconv_silu<<<eb(SS * d), 256, 0, s>>>(w.xdir, dp.conv_w, dp.conv_b,
                                             w.xc[dir], d);
    gemm(s, w.xc[dir], d, 0, dp.xp_w, nullptr, w.xdbl[dir], nx, 0, SS, nx, d, 0, w);
    gemm(s, w.xdbl[dir], nx, 0, dp.dt_w, dp.dt_b, w.dt[dir], d, 0, SS, d, r, 4, w);
  }
  ScanArgs sa;
  for (int dir = 0; dir < 2; ++dir) {
    const DirP& dp = dir ? m.bwd : m.fwd;
    sa.u[dir] = w.xc[dir]; sa.dt[dir] = w.dt[dir]; sa.xdbl[dir] = w.xdbl[dir];
    sa.Alog[dir] = dp.A_log; sa.Dp[dir] = dp.D; sa.y[dir] = w.ydir[dir];
  }
  k_scan<<<dim3(BN, 2), d, 0, s>>>(sa, w.xz, d2, d, nx, r, r + NST, d);
  k_scale_add<<<eb(SS * d), 256, 0, s>>>(w.ysum, w.ydir[0], w.ydir[1], SS * d, 0.5f);
  lnorm(s, w.ysum, m.ng, m.nb, w.ln, d);
  gemm(s, w.ln, d, 0, m.out_w, nullptr, out, d, 0, SS, d, d, 0, w);
}

void run_cross(hipStream_t s, const CrossP& p, const float* x0, const float* x1,
               float* out, WS& w) {
  lnorm(s, x0, p.n0g, p.n0b, w.n0, CD);
  lnorm(s, x1, p.n1g, p.n1b, w.n1, CD);
  k_slice<<<eb(SS * CD), 256, 0, s>>>(w.n0, CD, 0, w.comb, 2 * CD, 0,  CD, 0);
  k_slice<<<eb(SS * CD), 256, 0, s>>>(w.n1, CD, 0, w.comb, 2 * CD, CD, CD, 0);
  gemm(s, w.comb, 2 * CD, 0, p.cw_w, p.cw_b, w.t64a, CD, 0, SS, CD, 2 * CD, 0, w);
  lnorm(s, w.t64a, p.cwg, p.cwb, w.t64b, CD);
  k_sigclip<<<eb(SS * CD), 256, 0, s>>>(w.t64b, w.wgt, SS * CD);
  run_mamba(s, p.m, w.comb, 2 * CD, 0, 2 * CD, w.mo, w);
  lnorm(s, w.mo, p.png, p.pnb, w.ln, 2 * CD);
  gemm(s, w.ln, 2 * CD, 0, p.pb_w, p.pb_b, w.t64a, CD, 0, SS, CD, 2 * CD, 0, w);
  k_mix<<<eb(SS * CD), 256, 0, s>>>(out, w.t64a, w.wgt, w.n0, x0, SS * CD);
}

void run_single(hipStream_t s, const SingleP& p, float* x, WS& w) {
  lnorm(s, x, p.ng0, p.nb0, w.t64a, CD);
  run_mamba(s, p.m, w.t64a, CD, 0, CD, w.mo, w);
  lnorm(s, w.mo, p.png, p.pnb, w.t64b, CD);
  k_scale_add<<<eb(SS * CD), 256, 0, s>>>(x, x, w.t64b, SS * CD, 1.0f);
}

} // namespace

extern "C" void kernel_launch(void* const* d_in, const int* in_sizes, int n_in,
                              void* d_out, int out_size, void* d_ws, size_t ws_size,
                              hipStream_t stream) {
  (void)in_sizes; (void)n_in; (void)out_size; (void)ws_size;
  Cursor cur{d_in, 0};
  const float* img1  = cur.next();
  const float* img2  = cur.next();
  const float* img1s = cur.next();
  const float* img2s = cur.next();
  CrossP spa[2];  for (int i = 0; i < 2; ++i) spa[i] = read_cross(cur);
  CrossP spe[2];  for (int i = 0; i < 2; ++i) spe[i] = read_cross(cur);
  CrossP spaX = read_cross(cur);
  CrossP speX = read_cross(cur);
  SingleP selfA = read_single(cur);
  const float* out_ln_g = cur.next();
  const float* out_ln_b = cur.next();
  const float* out_w1   = cur.next();
  const float* out_b1   = cur.next();
  const float* out_w2   = cur.next();
  const float* out_b2   = cur.next();
  const float* alpha    = cur.next();
  const float* beta     = cur.next();

  float* wp = reinterpret_cast<float*>(d_ws);
  auto take = [&](size_t n) { float* p = wp; wp += n; return p; };
  float* seq1  = take((size_t)SS * CD);
  float* seq2  = take((size_t)SS * CD);
  float* seqs1 = take((size_t)SS * CD);   // reused as spa_fusion later
  float* seqs2 = take((size_t)SS * CD);   // reused as spe_fusion later
  WS w;
  w.n0   = take((size_t)SS * CD);
  w.n1   = take((size_t)SS * CD);
  w.comb = take((size_t)SS * 2 * CD);
  w.t64a = take((size_t)SS * CD);
  w.t64b = take((size_t)SS * CD);
  w.wgt  = take((size_t)SS * CD);
  w.xz   = take((size_t)SS * 4 * CD);
  w.xdir = take((size_t)SS * 2 * CD);
  w.ysum = take((size_t)SS * 2 * CD);
  w.ln   = take((size_t)SS * 2 * CD);
  w.mo   = take((size_t)SS * 2 * CD);
  w.fuse = take((size_t)SS * CD);
  for (int i = 0; i < 2; ++i) {
    w.xc[i]   = take((size_t)SS * 2 * CD);
    w.xdbl[i] = take((size_t)SS * 40);
    w.dt[i]   = take((size_t)SS * 2 * CD);
    w.ydir[i] = take((size_t)SS * 2 * CD);
  }
  w.Abf = reinterpret_cast<__bf16*>(take((size_t)SS * 128 / 2));  // M x Kp(max)
  w.Wbf = reinterpret_cast<__bf16*>(take((size_t)256 * 128 / 2)); // Np x Kp

  // ---- prep: nan_to_num + NCHW -> (B,L,C) ----
  k_prep<<<eb(BN * CD * LL), 256, 0, stream>>>(img1,  seq1);
  k_prep<<<eb(BN * CD * LL), 256, 0, stream>>>(img2,  seq2);
  k_prep<<<eb(BN * CD * LL), 256, 0, stream>>>(img1s, seqs1);
  k_prep<<<eb(BN * CD * LL), 256, 0, stream>>>(img2s, seqs2);

  // ---- depth loop ----
  for (int i = 0; i < 2; ++i) {
    run_cross(stream, spa[i], seq1, seqs1, seq1, w);
    run_cross(stream, spe[i], seq2, seqs2, seq2, w);
  }

  // ---- cross fusion (seq1/seq2 must stay intact) ----
  float* spaF = seqs1;
  float* speF = seqs2;
  run_cross(stream, spaX, seq1, seq2, spaF, w);
  run_cross(stream, speX, seq2, seq1, speF, w);
  k_fuse<<<eb(SS * CD), 256, 0, stream>>>(spaF, speF, alpha, beta, w.fuse, SS * CD);

  // ---- self block + output head ----
  run_single(stream, selfA, w.fuse, w);
  lnorm(stream, w.fuse, out_ln_g, out_ln_b, w.t64a, CD);
  gemm(stream, w.t64a, CD, 0, out_w1, out_b1, w.comb, 2 * CD, 0, SS, 2 * CD, CD, 3, w);
  gemm(stream, w.comb, 2 * CD, 0, out_w2, out_b2, w.t64b, CD, 0, SS, CD, 2 * CD, 0, w);
  k_final<<<eb(BN * CD * LL), 256, 0, stream>>>(w.t64b, seq1, seq2,
                                                reinterpret_cast<float*>(d_out));
}
